// SelfAttention_21723944583745
// MI455X (gfx1250) — matrix operand order repaired and checked by reading, not compiled, not run on hardware
//
#include <hip/hip_runtime.h>
#include <hip/hip_bf16.h>
#include <math.h>

typedef __attribute__((ext_vector_type(16))) _Float16 v16h;
typedef __attribute__((ext_vector_type(8)))  float    v8f;

#define SEQ 2048
#define DIM 1024
#define HDIM 64
#define NHEAD 16
#define NQKV 3072
#define INV_SQRT 0.125f

// workspace byte offsets
#define XH_OFF    (0u)          // x f16            [SEQ][DIM]       4 MB
#define WQKV_OFF  (4194304u)    // w_qkv f16        [NQKV][DIM]      6 MB
#define WOUT_OFF  (10485760u)   // w_out f16        [DIM][DIM]       2 MB
#define QH_OFF    (12582912u)   // q f16            [NH][SEQ][HD]    4 MB
#define KH_OFF    (16777216u)   // k f16            [NH][SEQ][HD]    4 MB
#define VT_OFF    (20971520u)   // v f16 transposed [NH][HD][SEQ]    4 MB
#define ATTN_OFF  (25165824u)   // attn f16         [SEQ][DIM]       4 MB

__device__ inline v8f wmma16(v16h a, v16h b, v8f c) {
  return __builtin_amdgcn_wmma_f32_16x16x32_f16(false, a, false, b, (short)0, c,
                                                false, false);
}

__device__ inline v8f vzero8() {
  v8f z = {0.f, 0.f, 0.f, 0.f, 0.f, 0.f, 0.f, 0.f};
  return z;
}

// A-matrix (16xK tile, 16-bit): lane l holds row M=l%16; halfs K..K+7 at
// base, K+16..K+23 at base+16 (lanes>=16 shifted by +8 in K).
__device__ inline v16h load_A(const _Float16* base, int ld, int row0, int k0) {
  const int l = threadIdx.x & 31;
  const _Float16* p =
      base + (size_t)(row0 + (l & 15)) * ld + k0 + ((l & 16) ? 8 : 0);
  union { uint4 u[2]; v16h h; } u;
  u.u[0] = *(const uint4*)p;
  u.u[1] = *(const uint4*)(p + 16);
  return u.h;
}

// B-matrix (Kx16 tile, 16-bit): lane l holds col N=l%16; halfs K..K+7 at
// base, K+16..K+23 at base+16 (lanes>=16 shifted by +8 in K). Source rows are N, cols are K.
__device__ inline v16h load_B(const _Float16* base, int ld, int n0, int k0) {
  const int l = threadIdx.x & 31;
  const _Float16* p =
      base + (size_t)(n0 + (l & 15)) * ld + k0 + ((l & 16) ? 8 : 0);
  union { uint4 u[2]; v16h h; } u;
  u.u[0] = *(const uint4*)p;
  u.u[1] = *(const uint4*)(p + 16);
  return u.h;
}

// Async DMA: each lane copies 16 B global -> LDS, tracked by ASYNCcnt.
__device__ inline void async_load_b128(const _Float16* gsrc, _Float16* ldst) {
  unsigned lofs = (unsigned)(uintptr_t)ldst;  // LDS aperture: low 32 = offset
  asm volatile("global_load_async_to_lds_b128 %0, %1, off"
               :: "v"(lofs), "v"(gsrc) : "memory");
}

__device__ inline float swz_xor8(float v) {
  union { float f; int i; } u;
  u.f = v;
  u.i = __builtin_amdgcn_ds_swizzle(u.i, (8 << 10) | 0x1f);
  return u.f;
}

__device__ inline float bmax16(float v) {
  union { float f; int i; } u, w;
  u.f = v;
  w.i = __builtin_amdgcn_ds_swizzle(u.i, (1 << 10) | 0x1f); u.f = fmaxf(u.f, w.f);
  w.i = __builtin_amdgcn_ds_swizzle(u.i, (2 << 10) | 0x1f); u.f = fmaxf(u.f, w.f);
  w.i = __builtin_amdgcn_ds_swizzle(u.i, (4 << 10) | 0x1f); u.f = fmaxf(u.f, w.f);
  w.i = __builtin_amdgcn_ds_swizzle(u.i, (8 << 10) | 0x1f); u.f = fmaxf(u.f, w.f);
  return u.f;
}

__device__ inline float bsum16(float v) {
  union { float f; int i; } u, w;
  u.f = v;
  w.i = __builtin_amdgcn_ds_swizzle(u.i, (1 << 10) | 0x1f); u.f += w.f;
  w.i = __builtin_amdgcn_ds_swizzle(u.i, (2 << 10) | 0x1f); u.f += w.f;
  w.i = __builtin_amdgcn_ds_swizzle(u.i, (4 << 10) | 0x1f); u.f += w.f;
  w.i = __builtin_amdgcn_ds_swizzle(u.i, (8 << 10) | 0x1f); u.f += w.f;
  return u.f;
}

// ---------------------------------------------------------------- convert
__global__ void convert_f16_kernel(const float* __restrict__ x,
                                   const float* __restrict__ wqkv,
                                   const float* __restrict__ wout,
                                   _Float16* __restrict__ xh,
                                   _Float16* __restrict__ wqkvh,
                                   _Float16* __restrict__ wouth) {
  int idx = blockIdx.x * blockDim.x + threadIdx.x;
  const int NX = SEQ * DIM;
  const int NW1 = NQKV * DIM;
  const int NW2 = DIM * DIM;
  if (idx < NX) { xh[idx] = (_Float16)x[idx]; return; }
  idx -= NX;
  if (idx < NW1) { wqkvh[idx] = (_Float16)wqkv[idx]; return; }
  idx -= NW1;
  if (idx < NW2) { wouth[idx] = (_Float16)wout[idx]; }
}

// ------------------------------------------------------- QKV GEMM + RoPE
// One wave per (16-row j tile, 64-col f tile). 128 * 48 = 6144 waves.
__global__ void qkv_rope_kernel(const _Float16* __restrict__ xh,
                                const _Float16* __restrict__ wqkvh,
                                const float* __restrict__ b_qkv,
                                _Float16* __restrict__ qh,
                                _Float16* __restrict__ kh,
                                _Float16* __restrict__ vT) {
  const int wave = (blockIdx.x * blockDim.x + threadIdx.x) >> 5;
  const int task = wave % 48;
  const int jt   = wave / 48;
  const int j0 = jt * 16;
  const int f0 = task * 64;
  const int ln = threadIdx.x & 31;
  const int n = ln & 15;
  const int rowofs = (ln & 16) ? 8 : 0;

  v8f acc[4];
#pragma unroll
  for (int nd = 0; nd < 4; ++nd) acc[nd] = vzero8();

  for (int kc = 0; kc < DIM; kc += 32) {
    v16h a = load_A(xh, DIM, j0, kc);
#pragma unroll
    for (int nd = 0; nd < 4; ++nd) {
      v16h b = load_B(wqkvh, DIM, f0 + nd * 16, kc);
      acc[nd] = wmma16(a, b, acc[nd]);
    }
  }

  if (f0 < 2048) {
    // q or k block: apply RoPE. Reference rotates across the HEAD axis
    // (reshape (S,HD,NH), split on last axis): partner head = h ^ 8, sign
    // -1 for h<8, +1 for h>=8; cos/sin depend on (j, d), d = f/16.
    _Float16* dst = (f0 < 1024) ? qh : kh;
    const int fb = f0 & 1023;
    const float sgn = (n < 8) ? -1.f : 1.f;
#pragma unroll
    for (int nd = 0; nd < 4; ++nd) {
      const int d = (fb >> 4) + nd;               // 0..63
      const float bq = b_qkv[f0 + nd * 16 + n];
      const float invf =
          __expf(-(float)(d & 31) * 0.28782313662425575f);  // ln(1e4)/32
#pragma unroll
      for (int r = 0; r < 8; ++r) {
        const int j = j0 + r + rowofs;
        float val = acc[nd][r] + bq;
        float part = swz_xor8(val);               // partner head h^8
        float ang = (float)j * invf;
        float c = cosf(ang), s = sinf(ang);
        float outv = val * c + sgn * part * s;
        dst[((size_t)n * SEQ + j) * HDIM + d] = (_Float16)outv;
      }
    }
  } else {
    // v block: store transposed [NH][HD][SEQ] for PV B-matrix loads
    const int fv = f0 - 2048;
#pragma unroll
    for (int nd = 0; nd < 4; ++nd) {
      const int d = (fv >> 4) + nd;
      const float bq = b_qkv[f0 + nd * 16 + n];
#pragma unroll
      for (int r = 0; r < 8; ++r) {
        const int j = j0 + r + rowofs;
        vT[((size_t)n * HDIM + d) * SEQ + j] = (_Float16)(acc[nd][r] + bq);
      }
    }
  }
}

// Stage one 32-key chunk of K (32x64 f16) and V (64x32 f16) into LDS with
// 16 async b128 DMA instructions (each moves 32 lanes x 16 B).
__device__ inline void stage_kv(const _Float16* __restrict__ khh,
                                const _Float16* __restrict__ vth,
                                _Float16* kb, _Float16* vb, int k0) {
  const int ln = threadIdx.x & 31;
  {
    const int sub = ln >> 3, seg = ln & 7;  // 4 rows x 8 segs per inst
#pragma unroll
    for (int i = 0; i < 8; ++i) {
      const int row = i * 4 + sub;
      async_load_b128(khh + (size_t)(k0 + row) * HDIM + seg * 8,
                      kb + row * HDIM + seg * 8);
    }
  }
  {
    const int vrow = ln >> 2, vseg = ln & 3;  // 8 rows x 4 segs per inst
#pragma unroll
    for (int i = 0; i < 8; ++i) {
      const int row = i * 8 + vrow;
      async_load_b128(vth + (size_t)row * SEQ + k0 + vseg * 8,
                      vb + row * 32 + vseg * 8);
    }
  }
}

// ------------------------------------------------ flash attention per head
// One wave per (head, 16-query tile): 16 * 128 = 2048 waves, 2 waves/block.
// Per-wave LDS: K dbl-buf 8 KB + V dbl-buf 8 KB + P stage 1 KB = 17 KB.
__global__ void attn_kernel(const float* __restrict__ bias,
                            const int* __restrict__ mask,
                            const _Float16* __restrict__ qh,
                            const _Float16* __restrict__ kh,
                            const _Float16* __restrict__ vT,
                            _Float16* __restrict__ attnh) {
  __shared__ _Float16 smem[2 * 8704];
  const int wid = threadIdx.x >> 5;
  const int wave = blockIdx.x * 2 + wid;
  const int h = wave & 15;
  const int jt = wave >> 4;
  const int j0 = jt * 16;
  const int ln = threadIdx.x & 31;
  const int n = ln & 15;
  const int rowofs = (ln & 16) ? 8 : 0;

  const _Float16* qhh = qh + (size_t)h * SEQ * HDIM;
  const _Float16* khh = kh + (size_t)h * SEQ * HDIM;
  const _Float16* vth = vT + (size_t)h * HDIM * SEQ;

  _Float16* kbase = &smem[wid * 8704];         // [2][32][64]
  _Float16* vbase = kbase + 4096;              // [2][64][32]
  _Float16* myp   = kbase + 8192;              // [16][32] P staging

  const v16h qa0 = load_A(qhh, HDIM, j0, 0);
  const v16h qa1 = load_A(qhh, HDIM, j0, 32);

  float m_run[8], l_run[8];
  v8f O[4];
#pragma unroll
  for (int r = 0; r < 8; ++r) { m_run[r] = -3.0e38f; l_run[r] = 0.f; }
#pragma unroll
  for (int nd = 0; nd < 4; ++nd) O[nd] = vzero8();

  stage_kv(khh, vth, kbase, vbase, 0);  // preload chunk 0 into buffer 0

  for (int k0 = 0; k0 < SEQ; k0 += 32) {
    const int b = (k0 >> 5) & 1;
    _Float16* kb = kbase + b * 2048;
    _Float16* vb = vbase + b * 2048;
    if (k0 + 32 < SEQ) {
      // WAR: previous iteration's ds reads of buffer b^1 must retire before
      // the DMA engine overwrites it.
      asm volatile("s_wait_dscnt 0" ::: "memory");
      stage_kv(khh, vth, kbase + (b ^ 1) * 2048, vbase + (b ^ 1) * 2048,
               k0 + 32);
      // 32 outstanding; async loads complete in order, so <=16 means the
      // current chunk's 16 transfers have landed in LDS.
      asm volatile("s_wait_asynccnt 16" ::: "memory");
    } else {
      asm volatile("s_wait_asynccnt 0" ::: "memory");
    }

    // scores: 16 queries x 32 keys, contraction over d=64 (K tile in LDS)
    v8f sC[2];
#pragma unroll
    for (int t = 0; t < 2; ++t) {
      v8f c = vzero8();
      c = wmma16(qa0, load_B(kb, HDIM, t * 16, 0), c);
      c = wmma16(qa1, load_B(kb, HDIM, t * 16, 32), c);
      sC[t] = c;
    }

    // bias + mask (dominant memory traffic of the whole problem)
    float sc[2][8];
#pragma unroll
    for (int t = 0; t < 2; ++t) {
      const int kc = k0 + t * 16 + n;
#pragma unroll
      for (int r = 0; r < 8; ++r) {
        const int j = j0 + r + rowofs;
        const size_t bidx = ((size_t)j * SEQ + kc) * NHEAD + h;
        const float bv = bias[bidx];
        const int mv = mask[bidx];
        const float s = fmaf(sC[t][r], INV_SQRT, bv);
        sc[t][r] = (mv == 0) ? -1.0e8f : s;
      }
    }
    if (k0 + 32 < SEQ) {  // speculative prefetch of next chunk
      const size_t pidx = ((size_t)j0 * SEQ + (k0 + 32 + n)) * NHEAD + h;
      __builtin_prefetch(&bias[pidx], 0, 0);
      __builtin_prefetch(&mask[pidx], 0, 0);
    }

    // online softmax (rows live in fixed lane-halves -> xor butterflies)
    float p[2][8];
#pragma unroll
    for (int r = 0; r < 8; ++r) {
      float v = fmaxf(sc[0][r], sc[1][r]);
      v = bmax16(v);
      const float m_new = fmaxf(m_run[r], v);
      const float alpha = __expf(m_run[r] - m_new);
      const float p0 = __expf(sc[0][r] - m_new);
      const float p1 = __expf(sc[1][r] - m_new);
      const float rs = bsum16(p0 + p1);
      l_run[r] = l_run[r] * alpha + rs;
      m_run[r] = m_new;
#pragma unroll
      for (int nd = 0; nd < 4; ++nd) O[nd][r] *= alpha;
      p[0][r] = p0;
      p[1][r] = p1;
    }

    // stage P (C layout) through wave-private LDS, reload in A layout
#pragma unroll
    for (int t = 0; t < 2; ++t)
#pragma unroll
      for (int r = 0; r < 8; ++r)
        myp[(r + rowofs) * 32 + t * 16 + n] = (_Float16)p[t][r];
    asm volatile("s_wait_dscnt 0" ::: "memory");
    const v16h ap = load_A(myp, 32, 0, 0);

    // O += P x V  (V tile in LDS, d-major [64][32])
#pragma unroll
    for (int nd = 0; nd < 4; ++nd) {
      v16h bv16 = load_B(vb, 32, nd * 16, 0);
      O[nd] = wmma16(ap, bv16, O[nd]);
    }
  }

  // normalize and store attn[j, d, h] as f16 feature-major (f = d*16 + h)
#pragma unroll
  for (int r = 0; r < 8; ++r) {
    const float inv = 1.0f / l_run[r];
    const int j = j0 + r + rowofs;
#pragma unroll
    for (int nd = 0; nd < 4; ++nd) {
      const int d = nd * 16 + n;
      attnh[(size_t)j * DIM + d * NHEAD + h] = (_Float16)(O[nd][r] * inv);
    }
  }
}

// ------------------------------------------------------------- out GEMM
// One wave per (16-row j tile, 64-col o tile): 128 * 16 = 2048 waves.
__global__ void out_gemm_kernel(const _Float16* __restrict__ attnh,
                                const _Float16* __restrict__ wouth,
                                const float* __restrict__ b_out,
                                float* __restrict__ out) {
  const int wave = (blockIdx.x * blockDim.x + threadIdx.x) >> 5;
  const int og = wave % 16;
  const int jt = wave / 16;
  const int j0 = jt * 16;
  const int o0 = og * 64;
  const int ln = threadIdx.x & 31;
  const int n = ln & 15;
  const int rowofs = (ln & 16) ? 8 : 0;

  v8f acc[4];
#pragma unroll
  for (int nd = 0; nd < 4; ++nd) acc[nd] = vzero8();

  for (int kc = 0; kc < DIM; kc += 32) {
    v16h a = load_A(attnh, DIM, j0, kc);
#pragma unroll
    for (int nd = 0; nd < 4; ++nd) {
      v16h b = load_B(wouth, DIM, o0 + nd * 16, kc);
      acc[nd] = wmma16(a, b, acc[nd]);
    }
  }
#pragma unroll
  for (int nd = 0; nd < 4; ++nd) {
    const float bo = b_out[o0 + nd * 16 + n];
#pragma unroll
    for (int r = 0; r < 8; ++r) {
      out[(size_t)(j0 + r + rowofs) * DIM + o0 + nd * 16 + n] =
          acc[nd][r] + bo;
    }
  }
}

extern "C" void kernel_launch(void* const* d_in, const int* in_sizes, int n_in,
                              void* d_out, int out_size, void* d_ws,
                              size_t ws_size, hipStream_t stream) {
  const float* x    = (const float*)d_in[0];
  const float* bias = (const float*)d_in[1];
  const int*   mask = (const int*)d_in[2];
  const float* wqkv = (const float*)d_in[3];
  const float* bqkv = (const float*)d_in[4];
  const float* wout = (const float*)d_in[5];
  const float* bout = (const float*)d_in[6];
  float* out = (float*)d_out;

  char* ws = (char*)d_ws;
  _Float16* xh    = (_Float16*)(ws + XH_OFF);
  _Float16* wqkvh = (_Float16*)(ws + WQKV_OFF);
  _Float16* wouth = (_Float16*)(ws + WOUT_OFF);
  _Float16* qh    = (_Float16*)(ws + QH_OFF);
  _Float16* kh    = (_Float16*)(ws + KH_OFF);
  _Float16* vT    = (_Float16*)(ws + VT_OFF);
  _Float16* attnh = (_Float16*)(ws + ATTN_OFF);

  const int nconv = SEQ * DIM + NQKV * DIM + DIM * DIM;  // 6,291,456
  convert_f16_kernel<<<(nconv + 255) / 256, 256, 0, stream>>>(
      x, wqkv, wout, xh, wqkvh, wouth);
  qkv_rope_kernel<<<768, 256, 0, stream>>>(xh, wqkvh, bqkv, qh, kh, vT);
  attn_kernel<<<1024, 64, 0, stream>>>(bias, mask, qh, kh, vT, attnh);
  out_gemm_kernel<<<256, 256, 0, stream>>>(attnh, wouth, bout, out);
}
